// VectorQuantizer_90752658965214
// MI455X (gfx1250) — matrix operand order repaired and checked by reading, not compiled, not run on hardware
//
#include <hip/hip_runtime.h>
#include <hip/hip_bf16.h>
#include <math.h>

#define N_EMBED 8192
#define C_DIM   256
#define HW      1024
#define M_PIX   16384          // 16*32*32
#define MT      64             // M tile per block
#define NT      128            // N chunk
#define LDB     264            // padded LDS stride (halves): 528B -> 4-bank rotation

typedef __attribute__((ext_vector_type(16))) __bf16 v16bf;
typedef __attribute__((ext_vector_type(8)))  float  v8f;

union FragBF { uint4 u[2]; v16bf v; };

// ---- workspace layout (bytes) ----
#define WS_ZHI   ((size_t)0)                    // 16384*256 bf16 = 8MB
#define WS_ZLO   ((size_t)8  << 20)             // 8MB
#define WS_EHI   ((size_t)16 << 20)             // 4MB
#define WS_ELO   ((size_t)20 << 20)             // 4MB
#define WS_ENORM (((size_t)24 << 20))           // 8192 f32
#define WS_IDX   (((size_t)24 << 20) + (32u<<10))   // 16384 i32
#define WS_CNT   (((size_t)24 << 20) + (96u<<10))   // 8192 u32
#define WS_LOSSP (((size_t)24 << 20) + (128u<<10))  // 16384 f32 partial losses

// ---------------- CDNA5 async copy helpers ----------------
__device__ __forceinline__ void async_b128(unsigned lds_off, const void* gaddr) {
    // GLOBAL_LOAD_ASYNC_TO_LDS_B128: VDST = LDS byte address (VGPR), VADDR = 64-bit global addr
    asm volatile("global_load_async_to_lds_b128 %0, %1, off"
                 :: "v"(lds_off), "v"((unsigned long long)(uintptr_t)gaddr)
                 : "memory");
}
__device__ __forceinline__ void wait_async0() {
    asm volatile("s_wait_asynccnt 0x0" ::: "memory");
}

// ---------------- embed prep: hi/lo split + ||e||^2 ----------------
__global__ void vq_embed_prep(const float* __restrict__ ew,
                              __bf16* __restrict__ ehi, __bf16* __restrict__ elo,
                              float* __restrict__ enorm) {
    __shared__ float red[256];
    const int n = blockIdx.x;
    const int c = threadIdx.x;
    float x = ew[n * C_DIM + c];
    __bf16 h = (__bf16)x;
    __bf16 l = (__bf16)(x - (float)h);
    ehi[n * C_DIM + c] = h;
    elo[n * C_DIM + c] = l;
    red[c] = x * x;
    __syncthreads();
    for (int s = 128; s > 0; s >>= 1) { if (c < s) red[c] += red[c + s]; __syncthreads(); }
    if (c == 0) enorm[n] = red[0];
}

// ---------------- z prep: NCHW -> NHWC transpose + hi/lo split ----------------
__global__ void vq_z_prep(const float* __restrict__ z,
                          __bf16* __restrict__ zhi, __bf16* __restrict__ zlo) {
    __shared__ float tile[32][33];
    const int blk = blockIdx.x;          // 16 * 8 * 32 = 4096 blocks
    const int b  = blk >> 8;
    const int ct = (blk >> 5) & 7;       // c-tile (8 of 32)
    const int ht = blk & 31;             // hw-tile (32 of 32)
    const int tx = threadIdx.x & 31;
    const int ty = threadIdx.x >> 5;     // 0..7
#pragma unroll
    for (int i = 0; i < 4; ++i) {        // coalesced over hw
        int cl = ty + i * 8;
        tile[cl][tx] = z[((size_t)(b * C_DIM + ct * 32 + cl)) * HW + ht * 32 + tx];
    }
    __syncthreads();
#pragma unroll
    for (int i = 0; i < 4; ++i) {        // coalesced over c
        int hwl = ty + i * 8;
        float x = tile[tx][hwl];
        __bf16 h = (__bf16)x;
        __bf16 l = (__bf16)(x - (float)h);
        size_t o = ((size_t)(b * HW + ht * 32 + hwl)) * C_DIM + ct * 32 + tx;
        zhi[o] = h; zlo[o] = l;
    }
}

// ---------------- main: fused (||e||^2 - 2 z.e) + argmin via bf16x3 WMMA ----------------
// A tile held entirely in registers; B double-buffered in LDS via async copies.
__global__ __launch_bounds__(256) void vq_argmin_wmma(
    const __bf16* __restrict__ zhi, const __bf16* __restrict__ zlo,
    const __bf16* __restrict__ ehi, const __bf16* __restrict__ elo,
    const float* __restrict__ enorm, int* __restrict__ idx_out) {
    extern __shared__ char smem[];
    const unsigned BUFH = NT * LDB;                 // halves per hi/lo array
    const unsigned smem_base = (unsigned)(size_t)smem;  // LDS byte offset (addr[31:0])

    const int tid  = threadIdx.x;
    const int lane = tid & 31;               // wave32
    const int wave = tid >> 5;               // 0..7
    const int rowg = wave >> 1;              // 16-row group (0..3)
    const int colg = wave & 1;               // 64-col group (0..1)
    const int m0   = blockIdx.x * MT;

    // ISA fragment layouts (wave32):
    // A 16x32 bf16: lane m(0..15) holds K {0..7,16..23}, lane m+16 holds K {8..15,24..31}
    const int aM    = lane & 15;
    const int aPart = (lane >> 4) << 3;      // 0 or 8
    // B 32x16: lane n(0..15) holds K {0..7,16..23}, lane n+16 holds K {8..15,24..31}
    const int bN    = lane & 15;
    const int bPart = (lane >> 4) << 3;      // 0 or 8

    // ---- A fragments (this wave's 16 rows x all 256 K, hi+lo) -> registers ----
    v16bf ahf[8], alf[8];
    {
        const __bf16* arow_hi = zhi + (size_t)(m0 + rowg * 16 + aM) * C_DIM;
        const __bf16* arow_lo = zlo + (size_t)(m0 + rowg * 16 + aM) * C_DIM;
#pragma unroll
        for (int k8 = 0; k8 < 8; ++k8) {
            FragBF fh, fl;
            fh.u[0] = *(const uint4*)(arow_hi + k8 * 32 + aPart);
            fh.u[1] = *(const uint4*)(arow_hi + k8 * 32 + 16 + aPart);
            fl.u[0] = *(const uint4*)(arow_lo + k8 * 32 + aPart);
            fl.u[1] = *(const uint4*)(arow_lo + k8 * 32 + 16 + aPart);
            ahf[k8] = fh.v; alf[k8] = fl.v;
        }
    }

    float ms[4][8];
    int   mi[4][8];
#pragma unroll
    for (int t = 0; t < 4; ++t)
#pragma unroll
        for (int r = 0; r < 8; ++r) { ms[t][r] = 3.4e38f; mi[t][r] = 0; }

    // ---- async stage of one B chunk (hi+lo) into LDS buffer at lds_base ----
    auto stage_b = [&](int nc, unsigned lds_base) {
#pragma unroll
        for (int j = 0; j < 32; ++j) {               // 8192 x 16B transfers / 256 threads
            int ch  = tid + j * 256;
            int arr = ch >> 12;                      // 0 = hi, 1 = lo
            int rem = ch & 4095;
            int r = rem >> 5, c16 = rem & 31;
            const __bf16* src = (arr ? elo : ehi) + (size_t)(nc + r) * C_DIM + c16 * 8;
            unsigned dst = lds_base + (unsigned)((arr * (int)BUFH + r * LDB + c16 * 8) * 2);
            async_b128(dst, src);
        }
    };

    // prologue: stage chunk 0 into buffer 0
    stage_b(0, smem_base);
    wait_async0();
    __syncthreads();

    for (int cidx = 0; cidx < N_EMBED / NT; ++cidx) {
        const int nc  = cidx * NT;
        const int cur = cidx & 1;
        if (nc + NT < N_EMBED)
            stage_b(nc + NT, smem_base + (unsigned)((cur ^ 1) * 2 * BUFH * 2));

        const __bf16* bHi = (const __bf16*)smem + (size_t)cur * 2 * BUFH;
        const __bf16* bLo = bHi + BUFH;

        v8f acc[4] = {};
#pragma unroll
        for (int k8 = 0; k8 < 8; ++k8) {
            const int kb = k8 * 32;
#pragma unroll
            for (int t = 0; t < 4; ++t) {
                FragBF bh, bl;
                const __bf16* pb  = bHi + (colg * 64 + t * 16 + bN) * LDB + kb + bPart;
                const __bf16* pbl = bLo + (colg * 64 + t * 16 + bN) * LDB + kb + bPart;
                bh.u[0] = *(const uint4*)pb;    bh.u[1] = *(const uint4*)(pb + 16);
                bl.u[0] = *(const uint4*)pbl;   bl.u[1] = *(const uint4*)(pbl + 16);
                // bf16x3: hi*hi + hi*lo + lo*hi, f32 accumulate
                acc[t] = __builtin_amdgcn_wmma_f32_16x16x32_bf16(false, ahf[k8], false, bh.v, (short)0, acc[t], false, false);
                acc[t] = __builtin_amdgcn_wmma_f32_16x16x32_bf16(false, ahf[k8], false, bl.v, (short)0, acc[t], false, false);
                acc[t] = __builtin_amdgcn_wmma_f32_16x16x32_bf16(false, alf[k8], false, bh.v, (short)0, acc[t], false, false);
            }
        }
        // epilogue: score = ||e_n||^2 - 2*dot ; running argmin per slot
#pragma unroll
        for (int t = 0; t < 4; ++t) {
            int n = nc + colg * 64 + t * 16 + bN;   // column n depends only on lane
            float en = enorm[n];
#pragma unroll
            for (int r = 0; r < 8; ++r) {
                float s = en - 2.0f * acc[t][r];
                if (s < ms[t][r]) { ms[t][r] = s; mi[t][r] = n; }
            }
        }
        wait_async0();       // next buffer fully in LDS
        __syncthreads();     // ...and visible to all waves; current buffer free to overwrite
    }

    // cross-lane/tile argmin reduction through LDS (reuse buffers)
    float* cs = (float*)smem;                       // [64][128]
    int*   ci = (int*)(smem + (size_t)MT * 128 * sizeof(float));
#pragma unroll
    for (int t = 0; t < 4; ++t) {
        int j = colg * 64 + t * 16 + (lane & 15);
#pragma unroll
        for (int r = 0; r < 8; ++r) {
            int row = rowg * 16 + r + ((lane >> 4) << 3);  // C/D layout: M = r + 8*(lane>=16)
            cs[row * 128 + j] = ms[t][r];
            ci[row * 128 + j] = mi[t][r];
        }
    }
    __syncthreads();
    if (tid < MT) {
        float best = 3.4e38f; int bi = 0;
        for (int j = 0; j < 128; ++j) {
            float s = cs[tid * 128 + j];
            if (s < best) { best = s; bi = ci[tid * 128 + j]; }
        }
        idx_out[m0 + tid] = bi;
    }
}

// ---------------- gather z_q, per-pixel loss partial, histogram ----------------
__global__ void vq_gather(const float* __restrict__ z, const float* __restrict__ ew,
                          const int* __restrict__ idx, float* __restrict__ out,
                          float* __restrict__ idx_f, unsigned int* __restrict__ counts,
                          float* __restrict__ lossp) {
    __shared__ float red[256];
    const int m = blockIdx.x;
    const int c = threadIdx.x;
    const int b = m >> 10, hw = m & 1023;
    const int id = idx[m];
    float e  = ew[(size_t)id * C_DIM + c];
    size_t zo = (size_t)b * C_DIM * HW + (size_t)c * HW + hw;
    float zv = z[zo];
    out[zo] = e;                         // straight-through forward value = z_q, NCHW
    float d = e - zv;
    red[c] = d * d;
    __syncthreads();
    for (int s = 128; s > 0; s >>= 1) { if (c < s) red[c] += red[c + s]; __syncthreads(); }
    if (c == 0) {
        lossp[m] = red[0];               // deterministic per-pixel partial
        idx_f[m] = (float)id;
        atomicAdd(&counts[id], 1u);      // integer atomics: order-independent
    }
}

// ---------------- finalize: loss + perplexity ----------------
__global__ void vq_finalize(const float* __restrict__ lossp,
                            const unsigned int* __restrict__ counts,
                            float* __restrict__ out_tail) {
    __shared__ float red[256];
    const int t = threadIdx.x;
    float ls = 0.f;
    for (int i = t; i < M_PIX; i += 256) ls += lossp[i];
    red[t] = ls; __syncthreads();
    for (int s = 128; s > 0; s >>= 1) { if (t < s) red[t] += red[t + s]; __syncthreads(); }
    float loss_total = red[0];
    __syncthreads();
    float ent = 0.f;
    for (int i = t; i < N_EMBED; i += 256) {
        float p = (float)counts[i] * (1.0f / (float)M_PIX);
        ent += p * logf(p + 1e-10f);
    }
    red[t] = ent; __syncthreads();
    for (int s = 128; s > 0; s >>= 1) { if (t < s) red[t] += red[t + s]; __syncthreads(); }
    if (t == 0) {
        out_tail[0] = 1.25f * loss_total / (float)(M_PIX * C_DIM);  // (BETA+1)*MSE
        out_tail[1] = expf(-red[0]);
    }
}

extern "C" void kernel_launch(void* const* d_in, const int* in_sizes, int n_in,
                              void* d_out, int out_size, void* d_ws, size_t ws_size,
                              hipStream_t stream) {
    const float* z  = (const float*)d_in[0];   // [16,256,32,32]
    const float* ew = (const float*)d_in[1];   // [8192,256]
    char* ws = (char*)d_ws;
    __bf16* zhi = (__bf16*)(ws + WS_ZHI);
    __bf16* zlo = (__bf16*)(ws + WS_ZLO);
    __bf16* ehi = (__bf16*)(ws + WS_EHI);
    __bf16* elo = (__bf16*)(ws + WS_ELO);
    float*  enorm = (float*)(ws + WS_ENORM);
    int*    idxw  = (int*)(ws + WS_IDX);
    unsigned int* counts = (unsigned int*)(ws + WS_CNT);
    float*  lossp = (float*)(ws + WS_LOSSP);

    float* out      = (float*)d_out;
    float* out_idx  = out + (size_t)16 * 256 * 32 * 32;   // 4,194,304
    float* out_tail = out_idx + M_PIX;                    // loss, perplexity

    hipMemsetAsync(counts, 0, N_EMBED * sizeof(unsigned int), stream);

    vq_embed_prep<<<N_EMBED, 256, 0, stream>>>(ew, ehi, elo, enorm);
    vq_z_prep<<<16 * 8 * 32, 256, 0, stream>>>(z, zhi, zlo);

    // LDS: 2 buffers x (hi+lo) x 128 rows x 264 halves x 2B = 270,336 B (< 320KB/WGP)
    size_t lds = (size_t)2 * 2 * NT * LDB * sizeof(__bf16);
    vq_argmin_wmma<<<M_PIX / MT, 256, lds, stream>>>(zhi, zlo, ehi, elo, enorm, idxw);

    vq_gather<<<M_PIX, 256, 0, stream>>>(z, ew, idxw, out, out_idx, counts, lossp);
    vq_finalize<<<1, 256, 0, stream>>>(lossp, counts, out_tail);
}
